// MGCL_50294067036842
// MI455X (gfx1250) — matrix operand order repaired and checked by reading, not compiled, not run on hardware
//
#include <hip/hip_runtime.h>
#include <hip/hip_bf16.h>
#include <math.h>

// ---------------------------------------------------------------------------
// MGCL forward for MI455X (gfx1250, wave32, WMMA).
// GEMMs use v_wmma_f32_16x16x32_bf16 (bf16 inputs, f32 accumulate); each wave
// produces a 16x32 output tile (two accumulators sharing one A fragment), and
// the whole block shares one LDS-staged bf16 B-slab per K-step (8x dedup).
// Teacher == student (identical params), biases feeding BN cancel exactly.
// ---------------------------------------------------------------------------

typedef __bf16 bf16;
typedef __attribute__((ext_vector_type(16))) __bf16 v16bf;
typedef __attribute__((ext_vector_type(8)))  float  v8f;

#define DF 128   // node feature dim
#define PHD 512  // predictor hidden
#define DDH 256  // decoder hidden

__device__ __forceinline__ float wave_sum(float v) {
#pragma unroll
  for (int off = 16; off > 0; off >>= 1) v += __shfl_xor(v, off, 32);
  return v;
}

// --------------------------- utility kernels -------------------------------

__global__ void k_fill(float* __restrict__ p, float v, size_t n) {
  size_t i = (size_t)blockIdx.x * blockDim.x + threadIdx.x;
  size_t stride = (size_t)gridDim.x * blockDim.x;
  for (; i < n; i += stride) p[i] = v;
}

__global__ void k_deg(const int* __restrict__ dst, float* __restrict__ deg, int E) {
  int e = blockIdx.x * blockDim.x + threadIdx.x;
  if (e < E) atomicAdd(&deg[dst[e]], 1.0f);
}

__global__ void k_rsqrt(float* __restrict__ p, int n) {
  int i = blockIdx.x * blockDim.x + threadIdx.x;
  if (i < n) p[i] = rsqrtf(p[i]);
}

// -------------------------- WMMA fragment helpers --------------------------
// A layout per lane l (16-bit A 16x32): row = l&15, half = l>>4;
// element e -> k0+half*16+e (contiguous, matches B).
__device__ __forceinline__ v16bf load_a_frag(const float* __restrict__ Arow,
                                             int k0, int half) {
  const float4* pa0 = (const float4*)(Arow + k0 + half * 16);
  const float4* pa1 = (const float4*)(Arow + k0 + 8 + half * 16);
  float4 q0 = pa0[0], q1 = pa0[1], q2 = pa1[0], q3 = pa1[1];
  v16bf a;
  a[0] = (bf16)q0.x; a[1] = (bf16)q0.y; a[2]  = (bf16)q0.z; a[3]  = (bf16)q0.w;
  a[4] = (bf16)q1.x; a[5] = (bf16)q1.y; a[6]  = (bf16)q1.z; a[7]  = (bf16)q1.w;
  a[8] = (bf16)q2.x; a[9] = (bf16)q2.y; a[10] = (bf16)q2.z; a[11] = (bf16)q2.w;
  a[12] = (bf16)q3.x; a[13] = (bf16)q3.y; a[14] = (bf16)q3.z; a[15] = (bf16)q3.w;
  return a;
}

// -------------------------- WMMA GEMM (bf16) -------------------------------
// C[M,Nc] = A[M,K] @ B[K,Nc] (+bias) (+relu). 16x32 output tile per wave;
// block stages the shared 32x32 B-slab (transposed, bf16) in LDS per K-step.
// B fragment per lane l: col = l&15(+16), k = k0 + (l>>4)*16 + e
//   -> contiguous LDS run Bs[col*32 + half*16 .. +15].

__global__ __launch_bounds__(256) void k_gemm_bf16(
    const float* __restrict__ A, const float* __restrict__ B,
    const float* __restrict__ bias, float* __restrict__ C,
    int M, int K, int Nc, int relu) {
  __shared__ alignas(32) bf16 Bs[32 * 32];
  const int tid = threadIdx.x;
  const int lane = tid & 31;
  const int wave = tid >> 5;
  int tileM = blockIdx.x * 128 + wave * 16;
  const bool active = tileM < M;
  if (!active) tileM = 0;                 // clamp: stay in loop for barriers
  const int tileN = blockIdx.y * 32;
  const int m = lane & 15;
  const int half = lane >> 4;
  const float* Arow = A + (size_t)(tileM + m) * K;
  const int col0 = tileN + m;
  const int col1 = tileN + 16 + m;
  const int sc = tid & 31;                // staging: column within slab
  const int sk4 = (tid >> 5) * 4;         // staging: 4 k-rows per thread
  v8f c0 = {}, c1 = {};
  for (int k0 = 0; k0 < K; k0 += 32) {
    // cooperative stage: B[k0..k0+31][tileN..tileN+31] -> Bs[c][k] (bf16)
#pragma unroll
    for (int j = 0; j < 4; ++j) {
      int kk = sk4 + j;
      Bs[sc * 32 + kk] = (bf16)B[(size_t)(k0 + kk) * Nc + tileN + sc];
    }
    __syncthreads();
    __builtin_prefetch(Arow + k0 + 64, 0, 0);
    v16bf a = load_a_frag(Arow, k0, half);
    v16bf b0 = *(const v16bf*)(&Bs[m * 32 + half * 16]);
    v16bf b1 = *(const v16bf*)(&Bs[(16 + m) * 32 + half * 16]);
    c0 = __builtin_amdgcn_wmma_f32_16x16x32_bf16(false, a, false, b0,
                                                 (short)0, c0, false, false);
    c1 = __builtin_amdgcn_wmma_f32_16x16x32_bf16(false, a, false, b1,
                                                 (short)0, c1, false, false);
    __syncthreads();
  }
  if (active) {
    const float bv0 = bias ? bias[col0] : 0.0f;
    const float bv1 = bias ? bias[col1] : 0.0f;
#pragma unroll
    for (int v = 0; v < 8; ++v) {
      const size_t row = (size_t)(tileM + v + half * 8);
      float v0 = c0[v] + bv0;
      float v1 = c1[v] + bv1;
      if (relu) { v0 = fmaxf(v0, 0.0f); v1 = fmaxf(v1, 0.0f); }
      C[row * Nc + col0] = v0;
      C[row * Nc + col1] = v1;
    }
  }
}

// Decoder GEMM: A row r is feature(r) = [e1+e2, e1*e2, e1, e2] (K=512) built
// on the fly from gathered S rows. Always bias+relu epilogue.
__global__ __launch_bounds__(256) void k_gemm_dec(
    const float* __restrict__ S, const int* __restrict__ inp,
    const float* __restrict__ B, const float* __restrict__ bias,
    float* __restrict__ C, int P, int Nc) {
  __shared__ alignas(32) bf16 Bs[32 * 32];
  const int K = 4 * DF;  // 512
  const int tid = threadIdx.x;
  const int lane = tid & 31;
  const int wave = tid >> 5;
  int tileM = blockIdx.x * 128 + wave * 16;
  const bool active = tileM < P;
  if (!active) tileM = 0;
  const int tileN = blockIdx.y * 32;
  const int m = lane & 15;
  const int half = lane >> 4;
  const int r = tileM + m;
  const int i0 = inp[r], i1 = inp[P + r];
  const float* s0 = S + (size_t)i0 * DF;
  const float* s1 = S + (size_t)i1 * DF;
  const int col0 = tileN + m;
  const int col1 = tileN + 16 + m;
  const int sc = tid & 31;
  const int sk4 = (tid >> 5) * 4;
  v8f c0 = {}, c1 = {};
  for (int k0 = 0; k0 < K; k0 += 32) {
#pragma unroll
    for (int j = 0; j < 4; ++j) {
      int kk = sk4 + j;
      Bs[sc * 32 + kk] = (bf16)B[(size_t)(k0 + kk) * Nc + tileN + sc];
    }
    __syncthreads();
    const int seg = k0 >> 7;            // 32-wide tile stays in one segment
    const int base = (k0 & 127) + half * 16;
    float f0[16], f1[16];
#pragma unroll
    for (int j = 0; j < 4; ++j) {
      int off = j * 4;  // 0,4,8,12
      float4 t0 = *(const float4*)(s0 + base + off);
      float4 t1 = *(const float4*)(s1 + base + off);
      f0[4 * j + 0] = t0.x; f0[4 * j + 1] = t0.y; f0[4 * j + 2] = t0.z; f0[4 * j + 3] = t0.w;
      f1[4 * j + 0] = t1.x; f1[4 * j + 1] = t1.y; f1[4 * j + 2] = t1.z; f1[4 * j + 3] = t1.w;
    }
    v16bf a;
#pragma unroll
    for (int e = 0; e < 16; ++e) {
      float v = (seg == 0) ? (f0[e] + f1[e])
              : (seg == 1) ? (f0[e] * f1[e])
              : (seg == 2) ? f0[e] : f1[e];
      a[e] = (bf16)v;
    }
    v16bf b0 = *(const v16bf*)(&Bs[m * 32 + half * 16]);
    v16bf b1 = *(const v16bf*)(&Bs[(16 + m) * 32 + half * 16]);
    c0 = __builtin_amdgcn_wmma_f32_16x16x32_bf16(false, a, false, b0,
                                                 (short)0, c0, false, false);
    c1 = __builtin_amdgcn_wmma_f32_16x16x32_bf16(false, a, false, b1,
                                                 (short)0, c1, false, false);
    __syncthreads();
  }
  if (active) {
    const float bv0 = bias[col0];
    const float bv1 = bias[col1];
#pragma unroll
    for (int v = 0; v < 8; ++v) {
      const size_t row = (size_t)(tileM + v + half * 8);
      C[row * Nc + col0] = fmaxf(c0[v] + bv0, 0.0f);
      C[row * Nc + col1] = fmaxf(c1[v] + bv1, 0.0f);
    }
  }
}

// -------------------- edge aggregation (D^-1/2 A D^-1/2) -------------------
// One wave per edge; each lane handles 4 contiguous channels. H fits in L2.
__global__ void k_agg(const float* __restrict__ H, const int* __restrict__ src,
                      const int* __restrict__ dst, const float* __restrict__ dinv,
                      float* __restrict__ OUT, int E, int total) {
  long long idx = (long long)blockIdx.x * blockDim.x + threadIdx.x;
  int e = (int)(idx >> 5);
  if (e >= total) return;
  int lane = (int)(idx & 31);
  int s, d;
  if (e < E) { s = src[e]; d = dst[e]; } else { s = e - E; d = s; }
  float nrm = dinv[s] * dinv[d];
  float4 h = *(const float4*)(H + (size_t)s * DF + lane * 4);
  float* o = OUT + (size_t)d * DF + lane * 4;
  atomicAdd(o + 0, h.x * nrm);
  atomicAdd(o + 1, h.y * nrm);
  atomicAdd(o + 2, h.z * nrm);
  atomicAdd(o + 3, h.w * nrm);
}

// ------------------------------- BatchNorm ---------------------------------
// blockDim.x == C; each thread owns one column over a row slab, coalesced.
__global__ void k_bnstats(const float* __restrict__ X, int Nrows, int C, int RPB,
                          float* __restrict__ sums, float* __restrict__ sumsq) {
  int c = threadIdx.x;
  int r0 = blockIdx.x * RPB;
  int r1 = min(r0 + RPB, Nrows);
  float s = 0.f, ss = 0.f;
  for (int r = r0; r < r1; ++r) {
    float v = X[(size_t)r * C + c];
    s += v; ss += v * v;
  }
  atomicAdd(&sums[c], s);
  atomicAdd(&sumsq[c], ss);
}

__global__ void k_bnfinal(const float* __restrict__ sums, const float* __restrict__ sumsq,
                          const float* __restrict__ g, const float* __restrict__ be,
                          float* __restrict__ scale, float* __restrict__ shift,
                          int C, float invN) {
  int c = threadIdx.x;
  if (c >= C) return;
  float mu = sums[c] * invN;
  float var = sumsq[c] * invN - mu * mu;
  float rstd = rsqrtf(var + 1e-5f);
  float sc = g[c] * rstd;
  scale[c] = sc;
  shift[c] = be[c] - mu * sc;
}

__global__ void k_bnprelu(float* __restrict__ X, size_t total, int cmask,
                          const float* __restrict__ scale, const float* __restrict__ shift,
                          const float* __restrict__ alpha) {
  size_t i = (size_t)blockIdx.x * blockDim.x + threadIdx.x;
  if (i >= total) return;
  int c = (int)(i & (size_t)cmask);
  float v = X[i] * scale[c] + shift[c];
  float a = alpha[0];
  X[i] = v > 0.f ? v : a * v;
}

// ------------------------- decoder tail & loss -----------------------------

__global__ void k_logred(const float* __restrict__ hidden, const float* __restrict__ w,
                         const float* __restrict__ b, float* __restrict__ out, int P) {
  int gid = blockIdx.x * (blockDim.x >> 5) + (threadIdx.x >> 5);
  int lane = threadIdx.x & 31;
  if (gid >= P) return;
  const float* row = hidden + (size_t)gid * DDH;
  float s = 0.f;
#pragma unroll
  for (int j = 0; j < 8; ++j) {
    int k = lane + 32 * j;
    s += row[k] * w[k];
  }
  s = wave_sum(s);
  if (lane == 0) out[gid] = s + b[0];
}

__global__ void k_rownorm(const float* __restrict__ X, float* __restrict__ rn, int Nrows) {
  int gid = blockIdx.x * (blockDim.x >> 5) + (threadIdx.x >> 5);
  int lane = threadIdx.x & 31;
  if (gid >= Nrows) return;
  const float* row = X + (size_t)gid * DF;
  float s = 0.f;
#pragma unroll
  for (int j = 0; j < 4; ++j) {
    float v = row[lane + 32 * j];
    s += v * v;
  }
  s = wave_sum(s);
  if (lane == 0) rn[gid] = rsqrtf(s + 1e-12f);
}

__global__ __launch_bounds__(256) void k_loss(
    const float* __restrict__ pred, const float* __restrict__ S,
    const float* __restrict__ rnp, const float* __restrict__ rnt,
    const int* __restrict__ ind, int Q, float* __restrict__ acc) {
  __shared__ float part[8];
  int wave = threadIdx.x >> 5;
  int lane = threadIdx.x & 31;
  int q = blockIdx.x * 8 + wave;
  float contrib = 0.f;
  if (q < Q) {
    int i0 = ind[q], i1 = ind[Q + q];
    const float* p0 = pred + (size_t)i0 * DF;
    const float* p1 = pred + (size_t)i1 * DF;
    const float* t0 = S + (size_t)i0 * DF;
    const float* t1 = S + (size_t)i1 * DF;
    float d1 = 0.f, d2 = 0.f;
#pragma unroll
    for (int j = 0; j < 4; ++j) {
      int k = lane + 32 * j;
      d1 += p0[k] * t1[k];
      d2 += p1[k] * t0[k];
    }
    d1 = wave_sum(d1);
    d2 = wave_sum(d2);
    contrib = 4.f - 2.f * (d1 * rnp[i0] * rnt[i1] + d2 * rnp[i1] * rnt[i0]);
  }
  if (lane == 0) part[wave] = contrib;
  __syncthreads();
  if (threadIdx.x == 0) {
    float s = 0.f;
#pragma unroll
    for (int w = 0; w < 8; ++w) s += part[w];
    atomicAdd(acc, s);
  }
}

__global__ void k_lossfin(const float* __restrict__ acc, float* __restrict__ out, float invQ) {
  out[0] = acc[0] * invQ;
}

// ------------------------------- launcher ----------------------------------

extern "C" void kernel_launch(void* const* d_in, const int* in_sizes, int n_in,
                              void* d_out, int out_size, void* d_ws, size_t ws_size,
                              hipStream_t stream) {
  const float* x   = (const float*)d_in[0];
  const int*   ei  = (const int*)d_in[1];
  const int*   inp = (const int*)d_in[2];
  const int*   ind = (const int*)d_in[3];
  const float* W1  = (const float*)d_in[4];
  const float* g1  = (const float*)d_in[6];
  const float* be1 = (const float*)d_in[7];
  const float* a1  = (const float*)d_in[8];
  const float* W2  = (const float*)d_in[9];
  const float* g2  = (const float*)d_in[11];
  const float* be2 = (const float*)d_in[12];
  const float* a2  = (const float*)d_in[13];
  const float* pW1 = (const float*)d_in[14];
  const float* pg  = (const float*)d_in[16];
  const float* pbe = (const float*)d_in[17];
  const float* pa  = (const float*)d_in[18];
  const float* pW2 = (const float*)d_in[19];
  const float* pb2 = (const float*)d_in[20];
  const float* dW1 = (const float*)d_in[21];
  const float* db1 = (const float*)d_in[22];
  const float* dW2 = (const float*)d_in[23];
  const float* db2 = (const float*)d_in[24];

  const int N = in_sizes[0] / DF;
  const int E = in_sizes[1] / 2;
  const int P = in_sizes[2] / 2;
  const int Q = in_sizes[3] / 2;
  const int total_edges = E + N;

  const int* e_src = ei;
  const int* e_dst = ei + E;

  // workspace layout (floats)
  float* W = (float*)d_ws;
  size_t off = 0;
  float* deg   = W + off; off += (size_t)N;
  float* sums  = W + off; off += 512;
  float* sumsq = W + off; off += 512;
  float* scale = W + off; off += 512;
  float* shift = W + off; off += 512;
  float* rnp   = W + off; off += (size_t)N;
  float* rnt   = W + off; off += (size_t)N;
  float* lacc  = W + off; off += 1;
  off = (off + 3) & ~(size_t)3;                 // keep 16B alignment below
  float* xw   = W + off; off += (size_t)N * DF; // X@W1 / H1@W2
  float* agg1 = W + off; off += (size_t)N * DF; // aggregated layer 1 -> H1
  float* agg2 = W + off; off += (size_t)N * DF; // aggregated layer 2 -> S
  float* big  = W + off; off += (size_t)N * PHD;// predictor hidden
  float* dech = xw;  // decoder hidden [P,256] reuses xw+agg1 (dead by then)

  float* out_pred = (float*)d_out;
  float* out_log  = out_pred + (size_t)N * DF;
  float* out_loss = out_log + P;

  auto fill = [&](float* p, float v, size_t n) {
    size_t b = (n + 255) / 256;
    if (b > 262144) b = 262144;
    k_fill<<<dim3((unsigned)b), dim3(256), 0, stream>>>(p, v, n);
  };
  auto gemm = [&](const float* A, const float* Bm, const float* bias, float* Cm,
                  int M, int K, int Nc, int relu) {
    dim3 g((M + 127) / 128, Nc / 32);
    k_gemm_bf16<<<g, dim3(256), 0, stream>>>(A, Bm, bias, Cm, M, K, Nc, relu);
  };

  // --- degrees & symmetric normalization ---
  fill(deg, 1.0f, (size_t)N);                     // self loops
  fill(agg1, 0.0f, (size_t)N * DF);
  fill(agg2, 0.0f, (size_t)N * DF);
  fill(lacc, 0.0f, 1);
  k_deg<<<dim3((E + 255) / 256), dim3(256), 0, stream>>>(e_dst, deg, E);
  k_rsqrt<<<dim3((N + 255) / 256), dim3(256), 0, stream>>>(deg, N);  // deg -> dinv

  const unsigned agg_blocks = (unsigned)(((long long)total_edges * 32 + 255) / 256);
  const unsigned ew_blocks128 = (unsigned)(((size_t)N * DF + 255) / 256);
  const unsigned ew_blocks512 = (unsigned)(((size_t)N * PHD + 255) / 256);

  // --- GCN layer 1: XW -> aggregate -> BN -> PReLU (b1 cancels in BN) ---
  gemm(x, W1, nullptr, xw, N, DF, DF, 0);
  k_agg<<<dim3(agg_blocks), dim3(256), 0, stream>>>(xw, e_src, e_dst, deg, agg1, E, total_edges);
  fill(sums, 0.0f, 1024);
  k_bnstats<<<dim3((N + 511) / 512), dim3(DF), 0, stream>>>(agg1, N, DF, 512, sums, sumsq);
  k_bnfinal<<<dim3(1), dim3(DF), 0, stream>>>(sums, sumsq, g1, be1, scale, shift, DF, 1.0f / N);
  k_bnprelu<<<dim3(ew_blocks128), dim3(256), 0, stream>>>(agg1, (size_t)N * DF, DF - 1, scale, shift, a1);

  // --- GCN layer 2 -> student embedding S (== teacher) ---
  gemm(agg1, W2, nullptr, xw, N, DF, DF, 0);
  k_agg<<<dim3(agg_blocks), dim3(256), 0, stream>>>(xw, e_src, e_dst, deg, agg2, E, total_edges);
  fill(sums, 0.0f, 1024);
  k_bnstats<<<dim3((N + 511) / 512), dim3(DF), 0, stream>>>(agg2, N, DF, 512, sums, sumsq);
  k_bnfinal<<<dim3(1), dim3(DF), 0, stream>>>(sums, sumsq, g2, be2, scale, shift, DF, 1.0f / N);
  k_bnprelu<<<dim3(ew_blocks128), dim3(256), 0, stream>>>(agg2, (size_t)N * DF, DF - 1, scale, shift, a2);

  // --- predictor: S@pW1 -> BN -> PReLU -> @pW2 + pb2 (pb1 cancels in BN) ---
  gemm(agg2, pW1, nullptr, big, N, DF, PHD, 0);
  fill(sums, 0.0f, 1024);
  k_bnstats<<<dim3((N + 511) / 512), dim3(PHD), 0, stream>>>(big, N, PHD, 512, sums, sumsq);
  k_bnfinal<<<dim3(1), dim3(PHD), 0, stream>>>(sums, sumsq, pg, pbe, scale, shift, PHD, 1.0f / N);
  k_bnprelu<<<dim3(ew_blocks512), dim3(256), 0, stream>>>(big, (size_t)N * PHD, PHD - 1, scale, shift, pa);
  gemm(big, pW2, pb2, out_pred, N, PHD, DF, 0);

  // --- decoder: on-the-fly feature GEMM (relu+bias) then 256->1 reduce ---
  {
    dim3 g((P + 127) / 128, DDH / 32);
    k_gemm_dec<<<g, dim3(256), 0, stream>>>(agg2, inp, dW1, db1, dech, P, DDH);
  }
  k_logred<<<dim3((P + 7) / 8), dim3(256), 0, stream>>>(dech, dW2, db2, out_log, P);

  // --- BYOL cosine loss ---
  k_rownorm<<<dim3((N + 7) / 8), dim3(256), 0, stream>>>(out_pred, rnp, N);
  k_rownorm<<<dim3((N + 7) / 8), dim3(256), 0, stream>>>(agg2, rnt, N);
  k_loss<<<dim3((Q + 7) / 8), dim3(256), 0, stream>>>(out_pred, agg2, rnp, rnt, ind, Q, lacc);
  k_lossfin<<<dim3(1), dim3(1), 0, stream>>>(lacc, out_loss, 1.0f / (float)Q);
}
